// ArcFaceInnerProduct_20100446945295
// MI455X (gfx1250) — compile-verified, matrix operand order checked by reading, not run on hardware
//
#include <hip/hip_runtime.h>
#include <hip/hip_bf16.h>

// ---- types for WMMA ----
typedef __attribute__((ext_vector_type(16))) _Float16 v16h;
typedef __attribute__((ext_vector_type(8)))  float    v8f;

#define BATCH 512
#define FDIM 512
#define CLS 100000
#define SCALE 30.0f

// ws layout (bytes):
//   [0..8)            : 2 float accumulators (sum feat norms, sum weight norms)
//   [64..400064)      : 100000 floats, reciprocal weight norms
//   [401408..925696)  : 512x512 f16 normalized features, PRE-TILED in WMMA
//                       fragment order: halfs[((mtile*16+kstep)*32+lane)*16+h]
#define WS_RNW_OFF_FLOATS 16
#define WS_NF16_OFF_BYTES 401408

#define STRF 516   // padded floats per LDS row (raw W tile)
#define STRH 520   // padded halfs per LDS row (f16 B tile)

__global__ __launch_bounds__(64)
void k_init(float* sums) {
    if (threadIdx.x < 2) sums[threadIdx.x] = 0.0f;
}

// Row norms of feat; emit normalized f16 A-matrix pre-tiled in WMMA fragment order.
__global__ __launch_bounds__(128)
void k_featnorm(const float* __restrict__ feat, _Float16* __restrict__ at,
                float* __restrict__ sums) {
    const int row = blockIdx.x;
    const int tid = threadIdx.x;
    const float* x = feat + (size_t)row * FDIM;
    float v[4];
    float p = 0.0f;
#pragma unroll
    for (int q = 0; q < 4; ++q) {
        float t = x[tid + q * 128];
        v[q] = t;
        p += t * t;
    }
    __shared__ float red[128];
    red[tid] = p;
    __syncthreads();
    for (int s = 64; s > 0; s >>= 1) {
        if (tid < s) red[tid] += red[tid + s];
        __syncthreads();
    }
    if (tid == 0) {
        float n = sqrtf(red[0]);
        atomicAdd(&sums[0], n);
        red[0] = 1.0f / n;
    }
    __syncthreads();
    const float inv = red[0];

    const int mtile = row >> 4;
    const int rlow  = row & 15;
#pragma unroll
    for (int q = 0; q < 4; ++q) {
        const int k     = tid + q * 128;
        const int kstep = k >> 5;
        const int kin   = k & 31;
        const int grp   = kin >> 3;
        const int hi    = kin & 7;
        const int lane  = rlow + ((grp & 1) << 4);
        const int h     = hi + ((grp >> 1) << 3);
        const size_t dst = ((size_t)(mtile * 16 + kstep) * 32 + lane) * 16 + h;
        at[dst] = (_Float16)(v[q] * inv);
    }
}

__global__ __launch_bounds__(256)
void k_wnorm(const float* __restrict__ W, float* __restrict__ rnw,
             float* __restrict__ sums) {
    const int wave = threadIdx.x >> 5;
    const int lane = threadIdx.x & 31;
    const int row = blockIdx.x * 8 + wave;            // 12500*8 = 100000 exactly
    const float4* w4 = (const float4*)(W + (size_t)row * FDIM);
    float p = 0.0f;
#pragma unroll
    for (int q = 0; q < 4; ++q) {
        float4 t = w4[lane + q * 32];
        p += t.x * t.x + t.y * t.y + t.z * t.z + t.w * t.w;
    }
#pragma unroll
    for (int off = 16; off > 0; off >>= 1) p += __shfl_xor(p, off, 32);
    if (lane == 0) {
        float n = sqrtf(p);
        atomicAdd(&sums[1], n);
        rnw[row] = 1.0f / n;
    }
}

// GEMM: cos = nfeat(f16,tiled) * (W/|W|)^T.
// Block: 16 classes x full 512 batch. B tile async-staged to LDS once,
// normalized+converted to f16 once, inner loop = ds_load_b128 + wmma with
// register double-buffered A fragments.
__global__ __launch_bounds__(256)
void k_gemm(const float* __restrict__ W, const _Float16* __restrict__ At,
            const float* __restrict__ rnw, float* __restrict__ out_cos,
            float* __restrict__ out_ml) {
    __shared__ __align__(16) float     shW[16 * STRF];   // ~33 KB raw W tile
    __shared__ __align__(16) _Float16  shB[16 * STRH];   // ~16.6 KB f16 B tile

    const int tid  = threadIdx.x;
    const int lane = tid & 31;
    const int wave = tid >> 5;
    const int c0   = blockIdx.x * 16;

    // ---- async-stage raw W tile (16 rows x 512 f32) into LDS ----
    {
        const float* gbase = W + (size_t)c0 * FDIM;
#pragma unroll
        for (int i = 0; i < 8; ++i) {
            const int idx = tid + i * 256;       // float4 index, 2048 total
            const int row = idx >> 7;
            const int c4  = idx & 127;
            const float* g = gbase + (size_t)row * FDIM + c4 * 4;
            unsigned ldsoff = (unsigned)(uintptr_t)(&shW[row * STRF + c4 * 4]);
            asm volatile("global_load_async_to_lds_b128 %0, %1, off"
                         :: "v"(ldsoff), "v"(g) : "memory");
        }
        asm volatile("s_wait_asynccnt 0" ::: "memory");
    }
    __syncthreads();

    // ---- scale by 1/|w| and convert to f16 (done once per tile) ----
#pragma unroll
    for (int i = 0; i < 8; ++i) {
        const int idx = tid + i * 256;
        const int row = idx >> 7;
        const int c4  = idx & 127;
        const float rw = rnw[c0 + row];
        const float4 f = *(const float4*)&shW[row * STRF + c4 * 4];
        union { _Float16 h[4]; uint2 u; } pk;
        pk.h[0] = (_Float16)(f.x * rw);
        pk.h[1] = (_Float16)(f.y * rw);
        pk.h[2] = (_Float16)(f.z * rw);
        pk.h[3] = (_Float16)(f.w * rw);
        *(uint2*)&shB[row * STRH + c4 * 4] = pk.u;
    }
    __syncthreads();

    // ---- main loop ----
    const int col = lane & 15;               // class within tile / M within A tile
    const int kb  = (lane >> 4) * 8;         // lanes 16-31 take the K+8 half
    const int cls = c0 + col;
    const int m0  = wave * 64;

    const v8f zero = {0.f, 0.f, 0.f, 0.f, 0.f, 0.f, 0.f, 0.f};
    v8f acc[4];
#pragma unroll
    for (int t = 0; t < 4; ++t) acc[t] = zero;

    union AV { uint4 u[2]; v16h v; };
    AV a[4], an[4];
    const _Float16* ab[4];
#pragma unroll
    for (int t = 0; t < 4; ++t) {
        // fragment base for tile ((m0/16)+t), kstep 0: contiguous 32 B / lane
        ab[t] = At + ((size_t)(((m0 >> 4) + t) * 16) * 32 + lane) * 16;
        a[t].u[0] = *(const uint4*)(ab[t]);
        a[t].u[1] = *(const uint4*)(ab[t] + 8);
    }

#pragma unroll
    for (int ks = 0; ks < 16; ++ks) {
        if (ks < 15) {
#pragma unroll
            for (int t = 0; t < 4; ++t) {
                const _Float16* p = ab[t] + (size_t)(ks + 1) * 512;
                an[t].u[0] = *(const uint4*)(p);
                an[t].u[1] = *(const uint4*)(p + 8);
            }
        }
        const _Float16* bp = &shB[col * STRH + ks * 32 + kb];
        union { uint4 u[2]; v16h v; } b;
        b.u[0] = *(const uint4*)(bp);
        b.u[1] = *(const uint4*)(bp + 16);
#pragma unroll
        for (int t = 0; t < 4; ++t)
            acc[t] = __builtin_amdgcn_wmma_f32_16x16x32_f16(
                false, a[t].v, false, b.v, (short)0, acc[t], false, false);
#pragma unroll
        for (int t = 0; t < 4; ++t) a[t] = an[t];
    }

    // ---- epilogue: cos and SCALE*cos ----
    const int mhi = (lane >> 4) * 8;
#pragma unroll
    for (int t = 0; t < 4; ++t) {
        const int mbase = m0 + t * 16 + mhi;
#pragma unroll
        for (int r = 0; r < 8; ++r) {
            const size_t idx = (size_t)(mbase + r) * CLS + cls;
            float c = acc[t][r];
            out_cos[idx] = c;
            out_ml[idx] = SCALE * c;
        }
    }
}

__global__ __launch_bounds__(512)
void k_fix(const int* __restrict__ label, const float* __restrict__ cosm,
           float* __restrict__ ml, float* __restrict__ scal,
           const float* __restrict__ sums) {
    const int tid = threadIdx.x;                 // 0..511 == batch row
    const int lab = label[tid];
    const size_t idx = (size_t)tid * CLS + lab;
    const float c = cosm[idx];
    const float cc = fminf(1.0f, fmaxf(-1.0f, c));
    const float theta = acosf(cc) * 57.29577951308232f;

    const float COSM = 0.8775825618903728f;      // cos(0.5)
    const float SINM = 0.479425538604203f;       // sin(0.5)
    const float THR  = -0.8775825618903728f;     // -cos(0.5)
    const float MSIN = 0.23971276930210156f;     // 0.5*sin(0.5)
    float nv;
    if (c > THR)
        nv = SCALE * (c * COSM - SINM * sqrtf(fmaxf(0.0f, 1.0f - c * c)));
    else
        nv = SCALE * (c - MSIN);
    ml[idx] = nv;

    __shared__ float ssum[512], ssq[512], smin[512], smax[512];
    ssum[tid] = theta; ssq[tid] = theta * theta;
    smin[tid] = theta; smax[tid] = theta;
    __syncthreads();
    for (int s = 256; s > 0; s >>= 1) {
        if (tid < s) {
            ssum[tid] += ssum[tid + s];
            ssq[tid]  += ssq[tid + s];
            smin[tid] = fminf(smin[tid], smin[tid + s]);
            smax[tid] = fmaxf(smax[tid], smax[tid + s]);
        }
        __syncthreads();
    }
    if (tid == 0) {
        const float sum = ssum[0], sq = ssq[0];
        const float avg = sum / 512.0f;
        const float var = (sq - sum * sum / 512.0f) / 511.0f;
        scal[0] = avg;                     // avg_theta
        scal[1] = smin[0];                 // min_theta
        scal[2] = smax[0];                 // max_theta
        scal[3] = sqrtf(fmaxf(var, 0.0f)); // stdv_theta
        scal[4] = sums[1] / (float)CLS;    // avg_w_norm
        scal[5] = sums[0] / (float)BATCH;  // avg_x_norm
    }
}

extern "C" void kernel_launch(void* const* d_in, const int* in_sizes, int n_in,
                              void* d_out, int out_size, void* d_ws, size_t ws_size,
                              hipStream_t stream) {
    const float* feat    = (const float*)d_in[0];
    const float* weights = (const float*)d_in[1];
    const int*   label   = (const int*)d_in[2];

    float* out  = (float*)d_out;
    float* cosm = out;                                  // [512 x 100000]
    float* ml   = out + (size_t)BATCH * CLS;            // [512 x 100000]
    float* scal = out + (size_t)2 * BATCH * CLS;        // 6 scalars

    float*     ws_f = (float*)d_ws;
    float*     rnw  = ws_f + WS_RNW_OFF_FLOATS;
    _Float16*  at   = (_Float16*)((char*)d_ws + WS_NF16_OFF_BYTES);

    k_init<<<1, 64, 0, stream>>>(ws_f);
    k_featnorm<<<BATCH, 128, 0, stream>>>(feat, at, ws_f);
    k_wnorm<<<CLS / 8, 256, 0, stream>>>(weights, rnw, ws_f);
    k_gemm<<<CLS / 16, 256, 0, stream>>>(weights, at, rnw, cosm, ml);
    k_fix<<<1, 512, 0, stream>>>(label, cosm, ml, scal, ws_f);
}
